// CustomConv2DPy_maskedConv_89515708383526
// MI455X (gfx1250) — matrix operand order repaired and checked by reading, not compiled, not run on hardware
//
#include <hip/hip_runtime.h>

// ---------------------------------------------------------------------------
// Masked ('A'-mask) 5x5 conv as implicit GEMM on CDNA5 WMMA (bf16 -> f32).
// B=4, Cin=192, Cout=384, H=W=128. Effective taps: (kh=0,kw=0..4),
// (kh=1,kw=0..4), (kh=2,kw=0..1)  -> 12 taps, K = 12*192 = 2304.
// Pipeline: double-buffered LDS; A-tile staged with gfx1250 async
// global->LDS b128 copies (ASYNCcnt), B-tile via coalesced transpose-gather.
// ---------------------------------------------------------------------------

typedef __attribute__((ext_vector_type(16))) __bf16 v16bf;
typedef __attribute__((ext_vector_type(8)))  __bf16 v8bf;
typedef __attribute__((ext_vector_type(8)))  float  v8f;
typedef int v4i __attribute__((vector_size(16)));   // matches builtin param pointee

union bfu { v16bf v; v8bf h[2]; };

#define B_   4
#define CIN  192
#define COUT 384
#define HW   128
#define KTOT 2304            // 12 taps * 192
#define KSTEPS 72            // 2304 / 32

#if defined(__HIP_DEVICE_COMPILE__) && __has_builtin(__builtin_amdgcn_global_load_async_to_lds_b128)
#define USE_ASYNC_LDS 1
#else
#define USE_ASYNC_LDS 0
#endif

#define AS_GLOBAL __attribute__((address_space(1)))
#define AS_LDS    __attribute__((address_space(3)))

static __device__ __forceinline__ void wait_async_zero() {
#if defined(__HIP_DEVICE_COMPILE__)
#if __has_builtin(__builtin_amdgcn_s_wait_asynccnt)
    __builtin_amdgcn_s_wait_asynccnt(0);
#else
    asm volatile("s_wait_asynccnt 0" ::: "memory");
#endif
#endif
}

static __device__ __forceinline__ unsigned short f2bf(float f) {
    union { float f; unsigned u; } v; v.f = f;
    unsigned u = v.u;
    unsigned r = (u + 0x7FFFu + ((u >> 16) & 1u)) >> 16;   // round-nearest-even
    return (unsigned short)r;
}

// ---- prep 1: x fp32 -> bf16 (vectorized float4 -> 4x bf16) ----------------
__global__ __launch_bounds__(256) void cvt_x_kernel(const float* __restrict__ x,
                                                    unsigned short* __restrict__ xbf) {
    const size_t i4 = (size_t)blockIdx.x * 256 + threadIdx.x;   // n/4 == 3145728 exact
    const float4 v = ((const float4*)x)[i4];
    uint2 p;
    p.x = (unsigned)f2bf(v.x) | ((unsigned)f2bf(v.y) << 16);
    p.y = (unsigned)f2bf(v.z) | ((unsigned)f2bf(v.w) << 16);
    ((uint2*)xbf)[i4] = p;
}

// ---- prep 2: masked weight -> bf16, layout [co][k], k = tap*192 + ci ------
__global__ __launch_bounds__(256) void prep_w_kernel(const float* __restrict__ wgt,
                                                     unsigned short* __restrict__ wbf) {
    const size_t gid = (size_t)blockIdx.x * 256 + threadIdx.x;  // 384*2304 = 884736 exact
    const int co  = (int)(gid / KTOT);
    const int k   = (int)(gid % KTOT);
    const int tap = k / CIN;
    const int ci  = k % CIN;
    const int kh  = (tap < 10) ? (tap / 5) : 2;
    const int kw  = (tap < 10) ? (tap % 5) : (tap - 10);
    const size_t src = ((size_t)(co * CIN + ci) * 5 + kh) * 5 + kw;
    wbf[gid] = f2bf(wgt[src]);
}

// ---- prep 3: valid[b][h][w] = any(mask != 0) in 5x5 window ----------------
__global__ __launch_bounds__(256) void prep_valid_kernel(const int* __restrict__ mask,
                                                         unsigned char* __restrict__ vld) {
    const int gid = blockIdx.x * 256 + threadIdx.x;             // 65536 exact
    const int b = gid >> 14, rem = gid & 16383;
    const int h = rem >> 7,  w = rem & 127;
    unsigned char v = 0;
    for (int dh = -2; dh <= 2; ++dh) {
        const int hh = h + dh;
        if (hh < 0 || hh >= HW) continue;
        for (int dw = -2; dw <= 2; ++dw) {
            const int ww = w + dw;
            if (ww < 0 || ww >= HW) continue;
            if (mask[(b << 14) + hh * HW + ww] != 0) v = 1;
        }
    }
    vld[gid] = v;
}

// ---- main: WMMA implicit GEMM ---------------------------------------------
// grid = (3, 512): x = 128-wide Cout tile, y = (b*128 + h) image row.
// block = 256 threads = 8 waves laid out 4(M) x 2(N); wave tile = 32x64.
__global__ __launch_bounds__(256) void masked_conv_wmma_kernel(
    const unsigned short* __restrict__ xbf,
    const unsigned short* __restrict__ wbf,
    const float* __restrict__ bias,
    const unsigned char* __restrict__ vld,
    float* __restrict__ out)
{
    __shared__ __align__(16) unsigned short As[2][128 * 32];  // [m][kk] 64B rows
    __shared__ __align__(16) unsigned short Bs[2][128 * 32];  // [n][kk] 64B rows

    const int tid    = threadIdx.x;
    const int wv     = tid >> 5;
    const int lane   = tid & 31;
    const int l15    = lane & 15;
    const int cobase = blockIdx.x * 128;
    const int row    = blockIdx.y;            // b*128 + h
    const int b      = row >> 7;
    const int h      = row & 127;

    const int wm = (wv & 3) * 32;             // wave M base
    const int wn = (wv >> 2) * 64;            // wave N base

    v8f acc[2][4];
#pragma unroll
    for (int mf = 0; mf < 2; ++mf)
#pragma unroll
        for (int nf = 0; nf < 4; ++nf)
            acc[mf][nf] = (v8f){0.f, 0.f, 0.f, 0.f, 0.f, 0.f, 0.f, 0.f};

    // staging thread mapping
    const int am    = tid >> 1;               // A: row m = 0..127
    const int ahalf = tid & 1;                //    elems [ahalf*16, +16)
    const int bn    = tid & 127;              // B: col n = 0..127
    const int bhalf = tid >> 7;               //    kk    [bhalf*16, +16)

    const int asel = (lane >> 4) * 8;         // A frag K-offset (0 or 8)
    const int bsel = (lane >> 4) * 8;         // B frag K-offset (0 or 8)

    // stage K-chunk kc into LDS buffer p
    auto stage = [&](int kc, int p) {
        const int k0  = kc * 32;              // == tap*192 + ci0
        const int tap = kc / 6;
        const int ci0 = (kc % 6) * 32;
        const int kh  = (tap < 10) ? (tap / 5) : 2;
        const int kw  = (tap < 10) ? (tap % 5) : (tap - 10);
        const int dh  = kh - 2, dw = kw - 2;

        // A (weights): contiguous, 16B aligned -> async global->LDS b128 x2
        {
            const unsigned short* src =
                wbf + (size_t)(cobase + am) * KTOT + k0 + ahalf * 16;
            unsigned short* dst = &As[p][am * 32 + ahalf * 16];
#if USE_ASYNC_LDS
            __builtin_amdgcn_global_load_async_to_lds_b128(
                (AS_GLOBAL v4i*)src, (AS_LDS v4i*)dst, 0, 0);
            __builtin_amdgcn_global_load_async_to_lds_b128(
                (AS_GLOBAL v4i*)src, (AS_LDS v4i*)dst, 16, 0);
#else
            const uint4 a0 = ((const uint4*)src)[0];
            const uint4 a1 = ((const uint4*)src)[1];
            uint4* d4 = (uint4*)dst;
            d4[0] = a0; d4[1] = a1;
#endif
        }
        // B (activations): transpose-gather 16 ci rows at fixed (h+dh, n+dw)
        {
            const int hsrc = h + dh;
            const int wsrc = bn + dw;
            union { uint4 q[2]; unsigned short s[16]; } pk;
            if (hsrc >= 0 && hsrc < HW && wsrc >= 0 && wsrc < HW) {
                const unsigned short* src =
                    xbf + (((size_t)(b * CIN + ci0 + bhalf * 16) * HW + hsrc) * HW + wsrc);
#pragma unroll
                for (int j = 0; j < 16; ++j) pk.s[j] = src[(size_t)j * (HW * HW)];
            } else {
                pk.q[0] = make_uint4(0, 0, 0, 0);
                pk.q[1] = make_uint4(0, 0, 0, 0);
            }
            uint4* dst = (uint4*)&Bs[p][bn * 32 + bhalf * 16];
            dst[0] = pk.q[0]; dst[1] = pk.q[1];
        }
        // prefetch the activation gather stream one more chunk ahead
        const int kc2 = kc + 1;
        if (kc2 < KSTEPS) {
            const int tap2 = kc2 / 6;
            const int ci2  = (kc2 % 6) * 32;
            const int kh2  = (tap2 < 10) ? (tap2 / 5) : 2;
            const int hs2  = h + kh2 - 2;
            if (hs2 >= 0 && hs2 < HW) {
                __builtin_prefetch(
                    xbf + (((size_t)(b * CIN + ci2 + bhalf * 16) * HW + hs2) * HW + bn),
                    0, 1);
            }
        }
    };

    // prologue: stage chunk 0 into buffer 0
    stage(0, 0);
    wait_async_zero();
    __syncthreads();

    for (int kc = 0; kc < KSTEPS; ++kc) {
        const int cur = kc & 1;
        if (kc + 1 < KSTEPS) stage(kc + 1, cur ^ 1);   // overlap with compute

        // ---- fragments per documented wave32 WMMA layouts ----------------
        v16bf afrag[2], bfrag[4];
#pragma unroll
        for (int mf = 0; mf < 2; ++mf) {
            const int m = wm + mf * 16 + l15;
            bfu u;
            u.h[0] = *(const v8bf*)&As[cur][m * 32 + asel];        // K 0..7  / 8..15
            u.h[1] = *(const v8bf*)&As[cur][m * 32 + asel + 16];   // K 16..23/ 24..31
            afrag[mf] = u.v;
        }
#pragma unroll
        for (int nf = 0; nf < 4; ++nf) {
            const int n = wn + nf * 16 + l15;
            bfu u;
            u.h[0] = *(const v8bf*)&Bs[cur][n * 32 + bsel];        // K 0..7  / 8..15
            u.h[1] = *(const v8bf*)&Bs[cur][n * 32 + bsel + 16];   // K 16..23/ 24..31
            bfrag[nf] = u.v;
        }
#pragma unroll
        for (int mf = 0; mf < 2; ++mf)
#pragma unroll
            for (int nf = 0; nf < 4; ++nf)
                acc[mf][nf] = __builtin_amdgcn_wmma_f32_16x16x32_bf16(
                    false, afrag[mf], false, bfrag[nf],
                    (short)0, acc[mf][nf], false, false);

        wait_async_zero();     // drain async LDS writes of next chunk
        __syncthreads();       // one barrier per K-step (double buffered)
    }

    // ---- epilogue: bias + valid gating, write all 128x128 tile elems ------
    const int mup = (lane >> 4) * 8;          // C/D layout: lanes 16..31 -> M+8
#pragma unroll
    for (int mf = 0; mf < 2; ++mf) {
#pragma unroll
        for (int nf = 0; nf < 4; ++nf) {
            const int n  = wn + nf * 16 + l15;                 // == w
            const unsigned char vm = vld[(size_t)row * HW + n];
#pragma unroll
            for (int r = 0; r < 8; ++r) {
                const int co = cobase + wm + mf * 16 + r + mup;
                const float val = vm ? (acc[mf][nf][r] + bias[co]) : 0.0f;
                out[(((size_t)b * COUT + co) * HW + h) * HW + n] = val;
            }
        }
    }
}

// ---- fallback (only if workspace is too small): direct fp32 conv ----------
__global__ __launch_bounds__(256) void fallback_conv_kernel(
    const float* __restrict__ x, const float* __restrict__ wgt,
    const float* __restrict__ bias, const int* __restrict__ mask,
    float* __restrict__ out)
{
    const size_t gid = (size_t)blockIdx.x * 256 + threadIdx.x;
    if (gid >= (size_t)B_ * COUT * HW * HW) return;
    const int w  = (int)(gid & 127);
    const int h  = (int)((gid >> 7) & 127);
    const int co = (int)((gid >> 14) % COUT);
    const int b  = (int)((gid >> 14) / COUT);

    unsigned char vm = 0;
    for (int dh = -2; dh <= 2; ++dh) {
        const int hh = h + dh; if (hh < 0 || hh >= HW) continue;
        for (int dw = -2; dw <= 2; ++dw) {
            const int ww = w + dw; if (ww < 0 || ww >= HW) continue;
            if (mask[(b << 14) + hh * HW + ww] != 0) vm = 1;
        }
    }
    if (!vm) { out[gid] = 0.0f; return; }

    float acc = bias[co];
    for (int kh = 0; kh < 3; ++kh) {
        const int kwlim = (kh == 2) ? 2 : 5;
        const int hh = h + kh - 2; if (hh < 0 || hh >= HW) continue;
        for (int kw = 0; kw < kwlim; ++kw) {
            const int ww = w + kw - 2; if (ww < 0 || ww >= HW) continue;
            for (int ci = 0; ci < CIN; ++ci) {
                acc += x[((size_t)(b * CIN + ci) * HW + hh) * HW + ww] *
                       wgt[((size_t)(co * CIN + ci) * 5 + kh) * 5 + kw];
            }
        }
    }
    out[gid] = acc;
}

extern "C" void kernel_launch(void* const* d_in, const int* in_sizes, int n_in,
                              void* d_out, int out_size, void* d_ws, size_t ws_size,
                              hipStream_t stream) {
    const float* x    = (const float*)d_in[0];
    const float* wgt  = (const float*)d_in[1];
    const float* bias = (const float*)d_in[2];
    const int*   mask = (const int*)d_in[3];
    float*       out  = (float*)d_out;

    const size_t XBF_BYTES = (size_t)B_ * CIN * HW * HW * 2;   // 25,165,824
    const size_t WBF_BYTES = (size_t)COUT * KTOT * 2;          //  1,769,472
    const size_t VAL_BYTES = (size_t)B_ * HW * HW;             //     65,536
    const size_t need = XBF_BYTES + WBF_BYTES + VAL_BYTES;

    if (ws_size < need) {
        const size_t n = (size_t)B_ * COUT * HW * HW;
        fallback_conv_kernel<<<dim3((unsigned)((n + 255) / 256)), dim3(256), 0, stream>>>(
            x, wgt, bias, mask, out);
        return;
    }

    unsigned short* xbf = (unsigned short*)d_ws;
    unsigned short* wbf = (unsigned short*)((char*)d_ws + XBF_BYTES);
    unsigned char*  vld = (unsigned char*)((char*)d_ws + XBF_BYTES + WBF_BYTES);

    cvt_x_kernel   <<<dim3(12288), dim3(256), 0, stream>>>(x, xbf);    // 12,582,912/4
    prep_w_kernel  <<<dim3(3456),  dim3(256), 0, stream>>>(wgt, wbf);  // 884,736
    prep_valid_kernel<<<dim3(256), dim3(256), 0, stream>>>(mask, vld); // 65,536
    masked_conv_wmma_kernel<<<dim3(3, 512), dim3(256), 0, stream>>>(xbf, wbf, bias, vld, out);
}
